// AndersonSolver_18614388261448
// MI455X (gfx1250) — matrix operand order repaired and checked by reading, not compiled, not run on hardware
//
#include <hip/hip_runtime.h>
#include <math.h>

// ---------------------------------------------------------------------------
// Anderson acceleration for x -> tanh(x @ W + b) on MI455X (gfx1250, wave32)
// B=256, H=2048, h = B*H = 524288, m=5 history, 20 iterations, beta=0.8
// GEMMs: v_wmma_f32_16x16x32_f16, fp32 accumulate. Global->LDS staging via
// the Tensor Data Mover (tensor_load_to_lds + s_wait_tensorcnt), double
// buffered. W is pre-transposed to f16 so A and B operand tiles share the
// same contiguous-K layout (all LDS fragment loads are wide/merged).
// ---------------------------------------------------------------------------

#define HTOT   524288      // flattened state length (B*H)
#define BB     256         // batch
#define KK     2048        // GEMM K
#define NNDIM  2048        // GEMM N
#define MHIST  5
#define NITER  20
#define BETA_C 0.8f
#define LAMBDA_C 1e-3f

typedef __attribute__((ext_vector_type(16))) _Float16 v16h;
typedef __attribute__((ext_vector_type(8)))  float    v8f;
typedef unsigned int v4u __attribute__((ext_vector_type(4)));
typedef int          v8i __attribute__((ext_vector_type(8)));
typedef int          v4i __attribute__((ext_vector_type(4)));

// ---------------- workspace layout (bytes) ----------------
#define OFF_X     ((size_t)0)
#define OFF_G     (OFF_X + (size_t)MHIST * HTOT * 4)
#define OFF_F     (OFF_G + (size_t)MHIST * HTOT * 4)
#define OFF_WT    (OFF_F + (size_t)MHIST * HTOT * 4)          // W^T as f16
#define OFF_XH    (OFF_WT + (size_t)KK * NNDIM * 2)
#define OFF_GRAMP (OFF_XH + (size_t)MHIST * HTOT * 2)
#define OFF_COEF  (OFF_GRAMP + (size_t)512 * 20 * 4)
#define OFF_NORMP (OFF_COEF + (size_t)64)

#define RED_BLOCKS 512

// ---------------------------------------------------------------------------
__device__ __forceinline__ float blockReduceSum256(float v, float* sdata) {
    const int tid = threadIdx.x;
    sdata[tid] = v;
    __syncthreads();
    #pragma unroll
    for (int s = 128; s > 0; s >>= 1) {
        if (tid < s) sdata[tid] += sdata[tid + s];
        __syncthreads();
    }
    float r = sdata[0];
    __syncthreads();
    return r;
}

// W (2048x2048 fp32, row-major k,n) -> Wt (2048x2048 f16, row-major n,k)
__global__ __launch_bounds__(256)
void aa_convertWt(const float* __restrict__ W, _Float16* __restrict__ Wt) {
    __shared__ float tile[32][33];
    const int bx = blockIdx.x * 32;     // n base
    const int by = blockIdx.y * 32;     // k base
    const int tx = threadIdx.x & 31;
    const int ty = threadIdx.x >> 5;    // 0..7
    #pragma unroll
    for (int r = 0; r < 32; r += 8)
        tile[ty + r][tx] = W[(size_t)(by + ty + r) * NNDIM + bx + tx];
    __syncthreads();
    #pragma unroll
    for (int r = 0; r < 32; r += 8)
        Wt[(size_t)(bx + ty + r) * KK + by + tx] = (_Float16)tile[tx][ty + r];
}

__global__ __launch_bounds__(256)
void aa_initX(const float* __restrict__ x0, float* __restrict__ X,
              _Float16* __restrict__ Xh) {
    size_t e = (size_t)blockIdx.x * 256 + threadIdx.x;
    if (e < (size_t)MHIST * HTOT) {
        float v = (e < (size_t)HTOT) ? x0[e] : 0.0f;
        X[e]  = v;
        Xh[e] = (_Float16)v;
    }
}

// ---------------------------------------------------------------------------
// GEMM + tanh:  out[M x 2048] = tanh(A[M x 2048] @ W + b),  M % 128 == 0.
// B operand supplied as Wt (N x K, contiguous K). Block tile 128x128,
// K-step 32 (one WMMA K), 8 wave32 waves. TDM stages both 128x32-half tiles
// (rows 64B data + 16B pad -> 80B stride, 16B-aligned, conflict-free banks)
// into double-buffered LDS; wave 0 drives the DMA.
// ---------------------------------------------------------------------------
#define GBM 128
#define GBN 128
#define GBK 32
#define ROW_B     80u                  // 64B row + 16B TDM pad
#define ROW_H     40                   // stride in halfs
#define TILE_BUF_B (128u * ROW_B)      // 10240
#define LDS_BYTES  (4u * TILE_BUF_B)   // 40960: A0,A1,B0,B1

__global__ __launch_bounds__(256)
void aa_gemm_tanh(const _Float16* __restrict__ A, const _Float16* __restrict__ Bt,
                  const float* __restrict__ bias, float* __restrict__ out, int M) {
    __shared__ __align__(16) char smem[LDS_BYTES];

    const int tid  = threadIdx.x;
    const int lane = tid & 31;
    const int wave = tid >> 5;          // 0..7
    const int wr   = wave & 3;          // rows wr*32 .. +31
    const int wc   = wave >> 2;         // cols wc*64 .. +63
    const int l16  = lane & 15;
    const int hl   = (lane >= 16) ? 1 : 0;

    const int bm0 = blockIdx.y * GBM;
    const int bn0 = blockIdx.x * GBN;

    v8f acc[2][4] = {};                 // 32x64 per wave = 2x4 tiles of 16x16

    // ---- TDM descriptors (block-uniform -> SGPRs) ----
    const unsigned lds_base = (unsigned)(uintptr_t)(&smem[0]);
    const unsigned ldsA0 = lds_base;
    const unsigned ldsA1 = lds_base + TILE_BUF_B;
    const unsigned ldsB0 = lds_base + 2u * TILE_BUF_B;
    const unsigned ldsB1 = lds_base + 3u * TILE_BUF_B;

    const unsigned long long aBase =
        (unsigned long long)(uintptr_t)A + (unsigned long long)bm0 * KK * 2ull;
    const unsigned long long bBase =
        (unsigned long long)(uintptr_t)Bt + (unsigned long long)bn0 * KK * 2ull;

    // Group1 (ISA 8.4): data_size=1(2B) [17:16], pad_en [20],
    // pad_interval=3 (64B) [24:22], pad_amount=3 (16B) [31:25].
    // tile 32 x 128, tensor_dim0 = 2048, dim0 stride = 2048.
    const v8i g1A = { (int)0x06D10000,
                      (int)(2048u << 16),                      // tensor_dim0 lo16
                      (int)(((unsigned)M & 0xFFFFu) << 16),    // tensor_dim1 = M
                      (int)(32u << 16),                        // tile_dim0 = 32
                      (int)128,                                // tile_dim1 = 128
                      (int)2048,                               // dim0 stride
                      0, 0 };
    const v8i g1B = { (int)0x06D10000,
                      (int)(2048u << 16),                      // tensor_dim0 = 2048
                      (int)(2048u << 16),                      // tensor_dim1 = 2048
                      (int)(32u << 16),                        // tile_dim0 = 32
                      (int)128,                                // tile_dim1 = 128
                      (int)2048,                               // dim0 stride
                      0, 0 };
    const v4i gz = {0, 0, 0, 0};

    auto issueTile = [&](unsigned lds, unsigned long long ga, v8i g1) {
        v4u g0;
        g0[0] = 1u;                                      // count=1, user mode
        g0[1] = lds;                                     // lds_addr
        g0[2] = (unsigned)(ga & 0xFFFFFFFFull);          // global_addr[31:0]
        g0[3] = (unsigned)((ga >> 32) & 0x01FFFFFFull)   // global_addr[56:32]
              | 0x80000000u;                             // type=2 ("image")
#if defined(__clang_major__) && __clang_major__ >= 23
        v8i gz8 = {0, 0, 0, 0, 0, 0, 0, 0};
        __builtin_amdgcn_tensor_load_to_lds(g0, g1, gz, gz, gz8, 0);
#else
        __builtin_amdgcn_tensor_load_to_lds(g0, g1, gz, gz, 0);
#endif
    };

    if (wave == 0) {                    // kick off step-0 tiles into buffer 0
        issueTile(ldsA0, aBase, g1A);
        issueTile(ldsB0, bBase, g1B);
    }

    const int nsteps = KK / GBK;        // 64
    for (int i = 0; i < nsteps; ++i) {
        const int cur = i & 1;
        if (wave == 0) {
            if (i + 1 < nsteps) {
                // prefetch next K-tiles into the other buffer, then wait for
                // the two oldest (current buffer) TDM ops: in-order per wave.
                const unsigned long long koff = (unsigned long long)(i + 1) * GBK * 2ull;
                issueTile(cur ? ldsA0 : ldsA1, aBase + koff, g1A);
                issueTile(cur ? ldsB0 : ldsB1, bBase + koff, g1B);
                __builtin_amdgcn_s_wait_tensorcnt(2);
            } else {
                __builtin_amdgcn_s_wait_tensorcnt(0);
            }
        }
        __syncthreads();   // buf[cur] now valid for all waves

        const _Float16* Asb = (const _Float16*)(smem + (cur ? TILE_BUF_B : 0u));
        const _Float16* Bsb = (const _Float16*)(smem + 2u * TILE_BUF_B
                                                     + (cur ? TILE_BUF_B : 0u));

        // A 16x32 f16 fragment: lane<16 -> K{0..15},
        // lane>=16 -> K{16..31}: contiguous 16-half run per lane (matches B).
        v16h afrag[2];
        #pragma unroll
        for (int tm = 0; tm < 2; ++tm) {
            const int base = (wr * 32 + tm * 16 + l16) * ROW_H + hl * 16;
            #pragma unroll
            for (int j = 0; j < 8; ++j)  afrag[tm][j]     = Asb[base + j];
            #pragma unroll
            for (int j = 0; j < 8; ++j)  afrag[tm][8 + j] = Asb[base + 8 + j];
        }
        // B 32x16 f16 fragment: N = lane%16; lane<16 -> K=j, lane>=16 -> K=16+j:
        // one contiguous 16-half run per lane (Bt rows are K-contiguous).
        v16h bfrag[4];
        #pragma unroll
        for (int tn = 0; tn < 4; ++tn) {
            const int base = (wc * 64 + tn * 16 + l16) * ROW_H + hl * 16;
            #pragma unroll
            for (int j = 0; j < 16; ++j) bfrag[tn][j] = Bsb[base + j];
        }

        #pragma unroll
        for (int tm = 0; tm < 2; ++tm)
            #pragma unroll
            for (int tn = 0; tn < 4; ++tn)
                acc[tm][tn] = __builtin_amdgcn_wmma_f32_16x16x32_f16(
                    false, afrag[tm], false, bfrag[tn],
                    (short)0, acc[tm][tn], false, false);

        __syncthreads();   // done reading buf[cur] before it is re-filled
    }

    // epilogue: tanh(acc + bias); C/D layout: M = r + 8*hl, N = lane%16
    #pragma unroll
    for (int tm = 0; tm < 2; ++tm) {
        #pragma unroll
        for (int tn = 0; tn < 4; ++tn) {
            const int gn = bn0 + wc * 64 + tn * 16 + l16;
            const float bv = bias[gn];
            #pragma unroll
            for (int r = 0; r < 8; ++r) {
                const int gm = bm0 + wr * 32 + tm * 16 + r + hl * 8;
                out[(size_t)gm * NNDIM + gn] = tanhf(acc[tm][tn][r] + bv);
            }
        }
    }
}

// ---------------------------------------------------------------------------
__global__ __launch_bounds__(256)
void aa_residual(const float* __restrict__ Grow, const float* __restrict__ Xrow,
                 float* __restrict__ Frow, float* __restrict__ normP) {
    __shared__ float sdata[256];
    float acc = 0.0f;
    const size_t stride = (size_t)gridDim.x * 256;
    for (size_t e = (size_t)blockIdx.x * 256 + threadIdx.x; e < (size_t)HTOT; e += stride) {
        float r = Grow[e] - Xrow[e];
        Frow[e] = r;
        acc += r * r;
    }
    float s = blockReduceSum256(acc, sdata);
    if (threadIdx.x == 0) normP[blockIdx.x] = s;
}

__global__ __launch_bounds__(256)
void aa_gram(const float* __restrict__ F, int kc, float* __restrict__ partials) {
    __shared__ float sdata[256];
    float acc[20];
    #pragma unroll
    for (int i = 0; i < 20; ++i) acc[i] = 0.0f;

    const size_t stride = (size_t)gridDim.x * 256;
    for (size_t e = (size_t)blockIdx.x * 256 + threadIdx.x; e < (size_t)HTOT; e += stride) {
        float f[MHIST];
        #pragma unroll
        for (int j = 0; j < MHIST; ++j) f[j] = F[(size_t)j * HTOT + e];
        float d[MHIST];
        #pragma unroll
        for (int j = 0; j < MHIST; ++j) d[j] = f[(j + 1) % MHIST] - f[j];
        const float fk = (kc == 0) ? f[0] : (kc == 1) ? f[1] : (kc == 2) ? f[2]
                        : (kc == 3) ? f[3] : f[4];
        int idx = 0;
        #pragma unroll
        for (int j = 0; j < MHIST; ++j)
            #pragma unroll
            for (int l = j; l < MHIST; ++l)
                acc[idx++] += d[j] * d[l];
        #pragma unroll
        for (int j = 0; j < MHIST; ++j) acc[15 + j] += d[j] * fk;
    }
    #pragma unroll
    for (int v = 0; v < 20; ++v) {
        float s = blockReduceSum256(acc[v], sdata);
        if (threadIdx.x == 0) partials[(size_t)blockIdx.x * 20 + v] = s;
    }
}

__global__ void aa_solve(const float* __restrict__ partials, int nPart, int kc,
                         float* __restrict__ coef) {
    __shared__ float sums[20];
    const int tid = threadIdx.x;
    if (tid < 20) {
        float s = 0.0f;
        for (int b = 0; b < nPart; ++b) s += partials[(size_t)b * 20 + tid];
        sums[tid] = s;
    }
    __syncthreads();
    if (tid == 0) {
        float Gm[MHIST][MHIST], c[MHIST];
        int idx = 0;
        for (int j = 0; j < MHIST; ++j)
            for (int l = j; l < MHIST; ++l) {
                Gm[j][l] = sums[idx];
                Gm[l][j] = sums[idx];
                ++idx;
            }
        for (int j = 0; j < MHIST; ++j) c[j] = sums[15 + j];

        float R[MHIST][MHIST];
        for (int i = 0; i < MHIST; ++i)
            for (int j = 0; j < MHIST; ++j) R[i][j] = 0.0f;
        for (int i = 0; i < MHIST; ++i) {
            float s = Gm[i][i];
            for (int t = 0; t < i; ++t) s -= R[t][i] * R[t][i];
            R[i][i] = sqrtf(fmaxf(s, 1e-20f));
            for (int j = i + 1; j < MHIST; ++j) {
                float v = Gm[i][j];
                for (int t = 0; t < i; ++t) v -= R[t][i] * R[t][j];
                R[i][j] = v / R[i][i];
            }
        }
        float y[MHIST];
        for (int i = 0; i < MHIST; ++i) {
            float v = c[i];
            for (int t = 0; t < i; ++t) v -= R[t][i] * y[t];
            y[i] = v / R[i][i];
        }
        float g[MHIST];
        for (int i = MHIST - 1; i >= 0; --i) {
            float v = y[i];
            for (int j = i + 1; j < MHIST; ++j) v -= R[i][j] * g[j];
            g[i] = v / (R[i][i] + LAMBDA_C);
        }
        for (int i = 0; i < MHIST; ++i) {
            float prev = g[(i + MHIST - 1) % MHIST];
            coef[i] = ((i == kc) ? 1.0f : 0.0f) + g[i] - prev;
        }
    }
}

__global__ __launch_bounds__(256)
void aa_update(const float* __restrict__ G, const float* __restrict__ F,
               const float* __restrict__ coef, float* __restrict__ Xrow,
               _Float16* __restrict__ Xhrow) {
    const float s0 = coef[0], s1 = coef[1], s2 = coef[2], s3 = coef[3], s4 = coef[4];
    const float omb = 1.0f - BETA_C;
    const size_t stride = (size_t)gridDim.x * 256;
    for (size_t e = (size_t)blockIdx.x * 256 + threadIdx.x; e < (size_t)HTOT; e += stride) {
        float x = s0 * (G[e]                    - omb * F[e]);
        x      += s1 * (G[(size_t)1 * HTOT + e] - omb * F[(size_t)1 * HTOT + e]);
        x      += s2 * (G[(size_t)2 * HTOT + e] - omb * F[(size_t)2 * HTOT + e]);
        x      += s3 * (G[(size_t)3 * HTOT + e] - omb * F[(size_t)3 * HTOT + e]);
        x      += s4 * (G[(size_t)4 * HTOT + e] - omb * F[(size_t)4 * HTOT + e]);
        Xrow[e]  = x;
        Xhrow[e] = (_Float16)x;
    }
}

__global__ void aa_norm_finalize(const float* __restrict__ normP, int n,
                                 float* __restrict__ dst) {
    __shared__ float sdata[256];
    float acc = 0.0f;
    for (int i = threadIdx.x; i < n; i += 256) acc += normP[i];
    float s = blockReduceSum256(acc, sdata);
    if (threadIdx.x == 0) *dst = sqrtf(s);
}

__global__ __launch_bounds__(256)
void aa_copyout(const float* __restrict__ src, float* __restrict__ dst) {
    const size_t stride = (size_t)gridDim.x * 256;
    for (size_t e = (size_t)blockIdx.x * 256 + threadIdx.x; e < (size_t)HTOT; e += stride)
        dst[e] = src[e];
}

// ---------------------------------------------------------------------------
extern "C" void kernel_launch(void* const* d_in, const int* in_sizes, int n_in,
                              void* d_out, int out_size, void* d_ws, size_t ws_size,
                              hipStream_t stream) {
    (void)in_sizes; (void)n_in; (void)out_size; (void)ws_size;

    const float* x0   = (const float*)d_in[0];
    const float* W    = (const float*)d_in[1];
    const float* bias = (const float*)d_in[2];
    float* out = (float*)d_out;

    char* ws = (char*)d_ws;
    float*    X     = (float*)(ws + OFF_X);
    float*    G     = (float*)(ws + OFF_G);
    float*    F     = (float*)(ws + OFF_F);
    _Float16* Wt    = (_Float16*)(ws + OFF_WT);
    _Float16* Xh    = (_Float16*)(ws + OFF_XH);
    float*    gramP = (float*)(ws + OFF_GRAMP);
    float*    coef  = (float*)(ws + OFF_COEF);
    float*    normP = (float*)(ws + OFF_NORMP);

    aa_convertWt<<<dim3(NNDIM / 32, KK / 32), 256, 0, stream>>>(W, Wt);
    aa_initX<<<(MHIST * HTOT) / 256, 256, 0, stream>>>(x0, X, Xh);

    // G = vmap(f)(X): one batched GEMM, M = 5*256 = 1280
    aa_gemm_tanh<<<dim3(NNDIM / GBN, (MHIST * BB) / GBM), 256, 0, stream>>>(
        Xh, Wt, bias, G, MHIST * BB);

    for (int j = 0; j < MHIST; ++j)
        aa_residual<<<RED_BLOCKS, 256, 0, stream>>>(G + (size_t)j * HTOT,
                                                    X + (size_t)j * HTOT,
                                                    F + (size_t)j * HTOT, normP);

    for (int k = 0; k < NITER; ++k) {
        const int kc = k % MHIST;
        const int kn = (k + 1) % MHIST;

        aa_gram<<<RED_BLOCKS, 256, 0, stream>>>(F, kc, gramP);
        aa_solve<<<1, 32, 0, stream>>>(gramP, RED_BLOCKS, kc, coef);
        aa_update<<<RED_BLOCKS, 256, 0, stream>>>(G, F, coef,
                                                  X + (size_t)kn * HTOT,
                                                  Xh + (size_t)kn * HTOT);
        aa_gemm_tanh<<<dim3(NNDIM / GBN, BB / GBM), 256, 0, stream>>>(
            Xh + (size_t)kn * HTOT, Wt, bias, G + (size_t)kn * HTOT, BB);
        aa_residual<<<RED_BLOCKS, 256, 0, stream>>>(G + (size_t)kn * HTOT,
                                                    X + (size_t)kn * HTOT,
                                                    F + (size_t)kn * HTOT, normP);
        aa_norm_finalize<<<1, 256, 0, stream>>>(normP, RED_BLOCKS, out + HTOT + k);
    }

    aa_copyout<<<RED_BLOCKS, 256, 0, stream>>>(X, out);
}
